// SimpleStableTransformer_22505628631688
// MI455X (gfx1250) — matrix operand order repaired and checked by reading, not compiled, not run on hardware
//
#include <hip/hip_runtime.h>
#include <hip/hip_bf16.h>
#include <math.h>

// ---------------- constants ----------------
constexpr int Bsz  = 4;
constexpr int Tsz  = 2048;
constexpr int Dsz  = 512;
constexpr int Hh   = 8;
constexpr int Ll   = 6;
constexpr int Dff  = 2048;
constexpr int Dk   = 64;            // Dsz / Hh
constexpr int Mrows = Bsz * Tsz;    // 8192
#define SCALEC 0.08838834764831845f /* 1/sqrt(2*Dk) */
#define INVT   0.66666666666666667f /* 1/1.5 */

typedef _Float16 half_t;
typedef __attribute__((ext_vector_type(16))) _Float16 v16h;
typedef __attribute__((ext_vector_type(8)))  _Float16 v8h;
typedef __attribute__((ext_vector_type(8)))  float    v8f;

// ---------------- helpers ----------------
__device__ inline float wave_reduce_sum(float v) {
  #pragma unroll
  for (int m = 16; m >= 1; m >>= 1) v += __shfl_xor(v, m, 32);
  return v;
}
__device__ inline float half_reduce_sum(float v) {   // within 16-lane halves
  #pragma unroll
  for (int m = 8; m >= 1; m >>= 1) v += __shfl_xor(v, m, 32);
  return v;
}
__device__ inline float half_reduce_max(float v) {
  #pragma unroll
  for (int m = 8; m >= 1; m >>= 1) v = fmaxf(v, __shfl_xor(v, m, 32));
  return v;
}

__device__ inline v8f wmma16(v16h a, v16h b, v8f c) {
  // v_wmma_f32_16x16x32_f16 : D = A(16x32) * B(32x16) + C
  return __builtin_amdgcn_wmma_f32_16x16x32_f16(false, a, false, b, (short)0, c, false, false);
}

// A fragment (16 rows x 32 K), src = &A[row0][k0], row stride ld (elements).
// A layout (matches B): lanes 0-15 -> rows, elems K0..15;
//                      lanes 16-31 -> rows,  elems K16..31.
__device__ inline v16h load_a_frag(const half_t* src, int ld, int lane) {
  int r  = lane & 15;
  int kk = (lane < 16) ? 0 : 16;
  const half_t* p = src + (size_t)r * ld + kk;
  union { v16h v; v8h h[2]; } u;
  u.h[0] = *(const v8h*)(p);
  u.h[1] = *(const v8h*)(p + 8);
  return u.v;
}

// B fragment (32 K x 16 cols) from an [N,K]-major array, src = &BT[col0][k0],
// row stride ld. Lanes 0-15: column n, K0..15; lanes 16-31: column n, K16..31.
__device__ inline v16h load_b_frag(const half_t* src, int ld, int lane) {
  int n  = lane & 15;
  int kk = (lane < 16) ? 0 : 16;
  return *(const v16h*)(src + (size_t)n * ld + kk);
}

// speculative prefetch (lowered to global_prefetch_b8; no counter traffic)
__device__ inline void pf(const void* p) { __builtin_prefetch(p, 0, 1); }

// ---------------- embedding + input LN ----------------
__global__ __launch_bounds__(256) void embed_kernel(
    const float* __restrict__ x, const float* __restrict__ ip_w,
    const float* __restrict__ ip_b, const float* __restrict__ g,
    const float* __restrict__ bb, const float* __restrict__ pos,
    const float* __restrict__ pad_tok, float* __restrict__ h)
{
  int lane = threadIdx.x & 31, wave = threadIdx.x >> 5;
  int row = blockIdx.x * 8 + wave;            // [0, B*T)
  int t = row % Tsz;
  float xv = x[row];
  xv = fminf(fmaxf(xv, -10.f), 10.f);
  bool pad = (xv == -1.0f);
  float v[16]; float s = 0.f, s2 = 0.f;
  #pragma unroll
  for (int i = 0; i < 16; i++) {
    int d = lane + i * 32;
    float val = xv * ip_w[d] + ip_b[d];
    v[i] = val; s += val; s2 += val * val;
  }
  s = wave_reduce_sum(s); s2 = wave_reduce_sum(s2);
  float mu = s * (1.f / Dsz);
  float var = s2 * (1.f / Dsz) - mu * mu;
  float rstd = rsqrtf(var + 1e-5f);
  #pragma unroll
  for (int i = 0; i < 16; i++) {
    int d = lane + i * 32;
    float val = (v[i] - mu) * rstd * g[d] + bb[d] + 0.1f * pos[(size_t)t * Dsz + d];
    if (pad) val = pad_tok[d];
    h[(size_t)row * Dsz + d] = val;
  }
}

// ---------------- layernorm (wave per row, D=512) ----------------
template <bool F16OUT>
__global__ __launch_bounds__(256) void ln_kernel(
    const float* __restrict__ in, const float* __restrict__ g,
    const float* __restrict__ bb, half_t* __restrict__ outH,
    float* __restrict__ outF)
{
  int lane = threadIdx.x & 31, wave = threadIdx.x >> 5;
  int row = blockIdx.x * 8 + wave;
  float v[16]; float s = 0.f, s2 = 0.f;
  #pragma unroll
  for (int i = 0; i < 16; i++) {
    int d = lane + i * 32;
    float val = in[(size_t)row * Dsz + d];
    v[i] = val; s += val; s2 += val * val;
  }
  s = wave_reduce_sum(s); s2 = wave_reduce_sum(s2);
  float mu = s * (1.f / Dsz);
  float var = s2 * (1.f / Dsz) - mu * mu;
  float rstd = rsqrtf(var + 1e-5f);
  #pragma unroll
  for (int i = 0; i < 16; i++) {
    int d = lane + i * 32;
    float val = (v[i] - mu) * rstd * g[d] + bb[d];
    if (F16OUT) outH[(size_t)row * Dsz + d] = (half_t)val;
    else        outF[(size_t)row * Dsz + d] = val;
  }
}

// ---------------- per-head L2 normalize (q,k) fp32 -> f16 ----------------
__global__ __launch_bounds__(256) void headnorm_kernel(
    const float* __restrict__ in, half_t* __restrict__ out)
{
  int lane = threadIdx.x & 31, wave = threadIdx.x >> 5;
  int unit = blockIdx.x * 8 + wave;           // [0, Mrows*H)
  int row = unit >> 3, hh = unit & 7;
  int d0 = hh * Dk + lane, d1 = d0 + 32;
  float a = in[(size_t)row * Dsz + d0];
  float b = in[(size_t)row * Dsz + d1];
  float ss = wave_reduce_sum(a * a + b * b);
  float inv = 1.0f / fmaxf(sqrtf(ss), 1e-8f);
  out[(size_t)row * Dsz + d0] = (half_t)(a * inv);
  out[(size_t)row * Dsz + d1] = (half_t)(b * inv);
}

// ---------------- V transpose: [B,T,D] fp32 -> [B*H, DK, T] f16 ----------------
__global__ __launch_bounds__(256) void vtrans_kernel(
    const float* __restrict__ v, half_t* __restrict__ vt)
{
  int o = blockIdx.x * 256 + threadIdx.x;     // [0, B*H*DK*T)
  int t = o & (Tsz - 1);
  int rest = o / Tsz;
  int dk = rest & (Dk - 1);
  int bh = rest / Dk;
  int b = bh >> 3, h = bh & 7;
  vt[o] = (half_t)v[((size_t)(b * Tsz + t)) * Dsz + h * Dk + dk];
}

// ---------------- weight convert+transpose: W[K,N] f32 -> WT[N,K] f16 ----------------
__global__ __launch_bounds__(256) void wconv_kernel(
    const float* __restrict__ w, half_t* __restrict__ wt, int K, int N)
{
  int o = blockIdx.x * 256 + threadIdx.x;     // [0, K*N)
  int k = o % K, n = o / K;
  wt[o] = (half_t)w[(size_t)k * N + n];
}

// ---------------- generic WMMA GEMM: C = A[M,K]f16 @ BT[N,K]f16^T + bias ---------
// MODE 0: store fp32    MODE 1: gelu(exact) -> f16
// MODE 2: CF += alpha*val   MODE 3: CF = clip(CF + alpha*val, +-10)
template <int MODE>
__global__ __launch_bounds__(256) void gemm_kernel(
    const half_t* __restrict__ A, const half_t* __restrict__ BT,
    const float* __restrict__ bias, float* __restrict__ CF,
    half_t* __restrict__ CH, int N, int K,
    const float* __restrict__ gate, int gidx, float gmul)
{
  int lane = threadIdx.x & 31, wave = threadIdx.x >> 5;
  int waveM = wave & 1, waveN = wave >> 1;        // 2 x 4 wave grid
  int rowBase = blockIdx.y * 128 + waveM * 64;    // 64 rows / wave
  int colBase = blockIdx.x * 128 + waveN * 32;    // 32 cols / wave
  v8f zero = {0.f,0.f,0.f,0.f,0.f,0.f,0.f,0.f};
  v8f c[4][2];
  #pragma unroll
  for (int i = 0; i < 4; i++)
    #pragma unroll
    for (int j = 0; j < 2; j++) c[i][j] = zero;

  for (int k0 = 0; k0 < K; k0 += 32) {
    // prefetch next K-step one iteration ahead (speculative, L2-resident data)
    if (k0 + 32 < K) {
      int r = lane & 15;
      pf(A + (size_t)(rowBase + r) * K + k0 + 32);
      pf(A + (size_t)(rowBase + 16 + r) * K + k0 + 32);
      pf(A + (size_t)(rowBase + 32 + r) * K + k0 + 32);
      pf(A + (size_t)(rowBase + 48 + r) * K + k0 + 32);
      pf(BT + (size_t)(colBase + r) * K + k0 + 32);
      pf(BT + (size_t)(colBase + 16 + r) * K + k0 + 32);
    }
    v16h a[4], b[2];
    #pragma unroll
    for (int i = 0; i < 4; i++)
      a[i] = load_a_frag(A + (size_t)(rowBase + i * 16) * K + k0, K, lane);
    #pragma unroll
    for (int j = 0; j < 2; j++)
      b[j] = load_b_frag(BT + (size_t)(colBase + j * 16) * K + k0, K, lane);
    #pragma unroll
    for (int i = 0; i < 4; i++)
      #pragma unroll
      for (int j = 0; j < 2; j++) c[i][j] = wmma16(a[i], b[j], c[i][j]);
  }

  float alpha = 1.f;
  if (MODE >= 2) alpha = gate[gidx] * gmul;
  int rOff = (lane >= 16) ? 8 : 0, cOff = lane & 15;
  #pragma unroll
  for (int j = 0; j < 2; j++) {
    int col = colBase + j * 16 + cOff;
    float bj = bias[col];
    #pragma unroll
    for (int i = 0; i < 4; i++) {
      int rb = rowBase + i * 16 + rOff;
      #pragma unroll
      for (int e = 0; e < 8; e++) {
        size_t idx = (size_t)(rb + e) * N + col;
        float val = c[i][j][e] + bj;
        if (MODE == 0) {
          CF[idx] = val;
        } else if (MODE == 1) {
          float gl = 0.5f * val * (1.0f + erff(val * 0.70710678118654752f));
          CH[idx] = (half_t)gl;
        } else if (MODE == 2) {
          CF[idx] += alpha * val;
        } else {
          float t = CF[idx] + alpha * val;
          CF[idx] = fminf(fmaxf(t, -10.f), 10.f);
        }
      }
    }
  }
}

// ---------------- flash attention: wave per 16-query tile per (b,h) --------------
__global__ __launch_bounds__(256) void attn_kernel(
    const half_t* __restrict__ q, const half_t* __restrict__ k,
    const half_t* __restrict__ vt, half_t* __restrict__ ao)
{
  __shared__ __align__(16) half_t pS[8][16][32];   // per-wave prob staging
  int lane = threadIdx.x & 31, wave = threadIdx.x >> 5;
  int gw = blockIdx.x * 8 + wave;                  // [0, B*H*(T/16))
  int qt = gw & (Tsz / 16 - 1);                    // & 127
  int bh = gw >> 7;
  int h = bh & 7, b = bh >> 3;
  int qbase = qt * 16;

  const half_t* qp = q + ((size_t)(b * Tsz + qbase)) * Dsz + h * Dk;
  const half_t* kp = k + (size_t)b * Tsz * Dsz + h * Dk;
  const half_t* vp = vt + (size_t)bh * Dk * Tsz;

  v16h qa0 = load_a_frag(qp, Dsz, lane);           // dk 0..31
  v16h qa1 = load_a_frag(qp + 32, Dsz, lane);      // dk 32..63

  float m_[8], l_[8];
  v8f zero = {0.f,0.f,0.f,0.f,0.f,0.f,0.f,0.f};
  v8f o_[4];
  #pragma unroll
  for (int e = 0; e < 8; e++) { m_[e] = -3e38f; l_[e] = 0.f; }
  #pragma unroll
  for (int j = 0; j < 4; j++) o_[j] = zero;

  int rOff = (lane >= 16) ? 8 : 0, cIdx = lane & 15;
  int kmax = qbase + 16;

  for (int kc = 0; kc < kmax; kc += 32) {
    // prefetch next key chunk (K rows + V rows), speculative
    if (kc + 32 < kmax) {
      pf(kp + (size_t)(kc + 32 + cIdx) * Dsz);
      pf(kp + (size_t)(kc + 48 + cIdx) * Dsz);
      pf(vp + (size_t)(lane & 31) * Tsz + kc + 32);
      pf(vp + (size_t)(32 + (lane & 31)) * Tsz + kc + 32);
    }
    // scores for keys [kc, kc+32)
    v8f s0 = zero, s1 = zero;
    {
      v16h kb;
      kb = load_b_frag(kp + (size_t)kc * Dsz, Dsz, lane);               s0 = wmma16(qa0, kb, s0);
      kb = load_b_frag(kp + (size_t)kc * Dsz + 32, Dsz, lane);          s0 = wmma16(qa1, kb, s0);
      kb = load_b_frag(kp + (size_t)(kc + 16) * Dsz, Dsz, lane);        s1 = wmma16(qa0, kb, s1);
      kb = load_b_frag(kp + (size_t)(kc + 16) * Dsz + 32, Dsz, lane);   s1 = wmma16(qa1, kb, s1);
    }
    float sc[8];
    #pragma unroll
    for (int e = 0; e < 8; e++) {
      int qrow = qbase + e + rOff;
      float v0 = fminf(fmaxf(s0[e] * SCALEC, -5.f), 5.f);
      if (kc + cIdx > qrow) v0 -= 10000.f;          // additive causal mask
      float v1 = fminf(fmaxf(s1[e] * SCALEC, -5.f), 5.f);
      if (kc + 16 + cIdx > qrow) v1 -= 10000.f;
      float rmax = half_reduce_max(fmaxf(v0, v1));
      float mnew = fmaxf(m_[e], rmax);
      float scale = __expf((m_[e] - mnew) * INVT);
      float p0 = __expf((v0 - mnew) * INVT);
      float p1 = __expf((v1 - mnew) * INVT);
      l_[e] = l_[e] * scale + half_reduce_sum(p0 + p1);
      m_[e] = mnew; sc[e] = scale;
      int r16 = e + rOff;
      pS[wave][r16][cIdx]      = (half_t)p0;
      pS[wave][r16][16 + cIdx] = (half_t)p1;
    }
    #pragma unroll
    for (int j = 0; j < 4; j++)
      #pragma unroll
      for (int e = 0; e < 8; e++) o_[j][e] *= sc[e];
    v16h pa = load_a_frag(&pS[wave][0][0], 32, lane);     // probs 16x32 (A layout)
    #pragma unroll
    for (int j = 0; j < 4; j++) {
      v16h vb = load_b_frag(vp + (size_t)(j * 16) * Tsz + kc, Tsz, lane);
      o_[j] = wmma16(pa, vb, o_[j]);
    }
  }

  #pragma unroll
  for (int j = 0; j < 4; j++)
    #pragma unroll
    for (int e = 0; e < 8; e++) {
      int qrow = qbase + e + rOff;
      float val = o_[j][e] / l_[e];
      ao[((size_t)(b * Tsz + qrow)) * Dsz + h * Dk + j * 16 + cIdx] = (half_t)val;
    }
}

// ---------------- masked mean pool over T ----------------
__global__ __launch_bounds__(256) void pool_kernel(
    const float* __restrict__ hf, const float* __restrict__ x,
    float* __restrict__ meanOut)
{
  int o = blockIdx.x * 256 + threadIdx.x;     // [0, B*D)
  int b = o >> 9, d = o & (Dsz - 1);
  float s = 0.f, c = 0.f;
  for (int t = 0; t < Tsz; t++) {
    float xv = fminf(fmaxf(x[b * Tsz + t], -10.f), 10.f);
    if (xv != -1.0f) { s += hf[((size_t)(b * Tsz + t)) * Dsz + d]; c += 1.f; }
  }
  meanOut[o] = s / fmaxf(c, 1.f);
}

// ---------------- output heads (wave per batch row) ----------------
__global__ __launch_bounds__(128) void heads_kernel(
    const float* __restrict__ mean,
    const float* bhg, const float* bhb, const float* bhw, const float* bhbias,
    const float* ahg, const float* ahb, const float* ahw, const float* ahbias,
    const float* chg, const float* chb, const float* chw, const float* chbias,
    float* __restrict__ out)
{
  int lane = threadIdx.x & 31, b = threadIdx.x >> 5;   // 4 waves
  float v[16]; float s = 0.f, s2 = 0.f;
  #pragma unroll
  for (int i = 0; i < 16; i++) {
    int d = lane + i * 32;
    float val = mean[b * Dsz + d];
    v[i] = val; s += val; s2 += val * val;
  }
  s = wave_reduce_sum(s); s2 = wave_reduce_sum(s2);
  float mu = s * (1.f / Dsz);
  float var = s2 * (1.f / Dsz) - mu * mu;
  float rstd = rsqrtf(var + 1e-5f);
  float d0 = 0.f, d1 = 0.f, da = 0.f, dc = 0.f;
  #pragma unroll
  for (int i = 0; i < 16; i++) {
    int d = lane + i * 32;
    float nb = (v[i] - mu) * rstd;
    float nbh = nb * bhg[d] + bhb[d];
    d0 += nbh * bhw[d * 2 + 0];
    d1 += nbh * bhw[d * 2 + 1];
    float nah = nb * ahg[d] + ahb[d];
    da += nah * ahw[d];
    float nch = nb * chg[d] + chb[d];
    dc += nch * chw[d];
  }
  d0 = wave_reduce_sum(d0); d1 = wave_reduce_sum(d1);
  da = wave_reduce_sum(da); dc = wave_reduce_sum(dc);
  if (lane == 0) {
    out[b * 4 + 0] = d0 + bhbias[0];
    out[b * 4 + 1] = d1 + bhbias[1];
    out[b * 4 + 2] = da + ahbias[0];
    float cv = dc + chbias[0];
    out[b * 4 + 3] = 1.f / (1.f + __expf(-cv));
  }
}

// ---------------- host orchestration ----------------
extern "C" void kernel_launch(void* const* d_in, const int* in_sizes, int n_in,
                              void* d_out, int out_size, void* d_ws, size_t ws_size,
                              hipStream_t stream) {
  (void)in_sizes; (void)n_in; (void)out_size; (void)ws_size;
  const float* x       = (const float*)d_in[0];
  const float* ip_w    = (const float*)d_in[1];
  const float* ip_b    = (const float*)d_in[2];
  const float* ip_ln_g = (const float*)d_in[3];
  const float* ip_ln_b = (const float*)d_in[4];
  const float* pos     = (const float*)d_in[5];
  const float* pad_tok = (const float*)d_in[6];
  const float* Wq = (const float*)d_in[7];  const float* bq = (const float*)d_in[8];
  const float* Wk = (const float*)d_in[9];  const float* bk = (const float*)d_in[10];
  const float* Wv = (const float*)d_in[11]; const float* bv = (const float*)d_in[12];
  const float* Wo = (const float*)d_in[13]; const float* bo = (const float*)d_in[14];
  const float* ln1g = (const float*)d_in[15]; const float* ln1b = (const float*)d_in[16];
  const float* ln2g = (const float*)d_in[17]; const float* ln2b = (const float*)d_in[18];
  const float* W1 = (const float*)d_in[19]; const float* b1 = (const float*)d_in[20];
  const float* W2 = (const float*)d_in[21]; const float* b2 = (const float*)d_in[22];
  const float* gate1 = (const float*)d_in[23]; const float* gate2 = (const float*)d_in[24];
  const float* fng = (const float*)d_in[25]; const float* fnb = (const float*)d_in[26];
  const float* bhg = (const float*)d_in[27]; const float* bhb = (const float*)d_in[28];
  const float* ahg = (const float*)d_in[29]; const float* ahb = (const float*)d_in[30];
  const float* chg = (const float*)d_in[31]; const float* chb = (const float*)d_in[32];
  const float* bhw = (const float*)d_in[33]; const float* bhbias = (const float*)d_in[34];
  const float* ahw = (const float*)d_in[35]; const float* ahbias = (const float*)d_in[36];
  const float* chw = (const float*)d_in[37]; const float* chbias = (const float*)d_in[38];
  float* outp = (float*)d_out;

  // workspace carve (256B aligned)
  char* base = (char*)d_ws;
  size_t off = 0;
  auto carve = [&](size_t bytes) -> char* {
    char* p = base + off;
    off = (off + bytes + 255) & ~(size_t)255;
    return p;
  };
  half_t* wqT  = (half_t*)carve((size_t)Dsz * Dsz * 2);
  half_t* wkT  = (half_t*)carve((size_t)Dsz * Dsz * 2);
  half_t* wvT  = (half_t*)carve((size_t)Dsz * Dsz * 2);
  half_t* woT  = (half_t*)carve((size_t)Dsz * Dsz * 2);
  half_t* w1T  = (half_t*)carve((size_t)Dsz * Dff * 2);
  half_t* w2T  = (half_t*)carve((size_t)Dff * Dsz * 2);
  float*  hbuf = (float*) carve((size_t)Mrows * Dsz * 4);
  half_t* hn16 = (half_t*)carve((size_t)Mrows * Dsz * 2);
  half_t* q16  = (half_t*)carve((size_t)Mrows * Dsz * 2);
  half_t* k16  = (half_t*)carve((size_t)Mrows * Dsz * 2);
  half_t* vT16 = (half_t*)carve((size_t)Mrows * Dsz * 2);
  half_t* ao16 = (half_t*)carve((size_t)Mrows * Dsz * 2);
  half_t* mid16= (half_t*)carve((size_t)Mrows * Dff * 2);
  float*  tmp  = (float*) carve((size_t)Mrows * Dsz * 4);
  float*  meanB= (float*) carve((size_t)Bsz * Dsz * 4);

  dim3 blk(256);
  dim3 gemmD(Dsz / 128, Mrows / 128);   // N=512
  dim3 gemmF(Dff / 128, Mrows / 128);   // N=2048

  // embedding
  embed_kernel<<<Mrows / 8, blk, 0, stream>>>(x, ip_w, ip_b, ip_ln_g, ip_ln_b,
                                              pos, pad_tok, hbuf);

  for (int i = 0; i < Ll; i++) {
    // per-layer weight convert + transpose -> f16 [N,K]
    wconv_kernel<<<(Dsz * Dsz) / 256, blk, 0, stream>>>(Wq + (size_t)i * Dsz * Dsz, wqT, Dsz, Dsz);
    wconv_kernel<<<(Dsz * Dsz) / 256, blk, 0, stream>>>(Wk + (size_t)i * Dsz * Dsz, wkT, Dsz, Dsz);
    wconv_kernel<<<(Dsz * Dsz) / 256, blk, 0, stream>>>(Wv + (size_t)i * Dsz * Dsz, wvT, Dsz, Dsz);
    wconv_kernel<<<(Dsz * Dsz) / 256, blk, 0, stream>>>(Wo + (size_t)i * Dsz * Dsz, woT, Dsz, Dsz);
    wconv_kernel<<<(Dsz * Dff) / 256, blk, 0, stream>>>(W1 + (size_t)i * Dsz * Dff, w1T, Dsz, Dff);
    wconv_kernel<<<(Dff * Dsz) / 256, blk, 0, stream>>>(W2 + (size_t)i * Dff * Dsz, w2T, Dff, Dsz);

    // LN1 -> f16
    ln_kernel<true><<<Mrows / 8, blk, 0, stream>>>(hbuf, ln1g + i * Dsz, ln1b + i * Dsz,
                                                   hn16, nullptr);
    // Q
    gemm_kernel<0><<<gemmD, blk, 0, stream>>>(hn16, wqT, bq + i * Dsz, tmp, nullptr,
                                              Dsz, Dsz, nullptr, 0, 0.f);
    headnorm_kernel<<<(Mrows * Hh) / 8, blk, 0, stream>>>(tmp, q16);
    // K
    gemm_kernel<0><<<gemmD, blk, 0, stream>>>(hn16, wkT, bk + i * Dsz, tmp, nullptr,
                                              Dsz, Dsz, nullptr, 0, 0.f);
    headnorm_kernel<<<(Mrows * Hh) / 8, blk, 0, stream>>>(tmp, k16);
    // V (+ transpose to [B*H, DK, T])
    gemm_kernel<0><<<gemmD, blk, 0, stream>>>(hn16, wvT, bv + i * Dsz, tmp, nullptr,
                                              Dsz, Dsz, nullptr, 0, 0.f);
    vtrans_kernel<<<((size_t)Mrows * Dsz) / 256, blk, 0, stream>>>(tmp, vT16);
    // attention
    attn_kernel<<<(Bsz * Hh * (Tsz / 16)) / 8, blk, 0, stream>>>(q16, k16, vT16, ao16);
    // O projection + gated residual (gate1*0.5)
    gemm_kernel<2><<<gemmD, blk, 0, stream>>>(ao16, woT, bo + i * Dsz, hbuf, nullptr,
                                              Dsz, Dsz, gate1, i, 0.5f);
    // LN2 -> f16
    ln_kernel<true><<<Mrows / 8, blk, 0, stream>>>(hbuf, ln2g + i * Dsz, ln2b + i * Dsz,
                                                   hn16, nullptr);
    // FFN1 + exact GELU -> f16
    gemm_kernel<1><<<gemmF, blk, 0, stream>>>(hn16, w1T, b1 + i * Dff, nullptr, mid16,
                                              Dff, Dsz, nullptr, 0, 0.f);
    // FFN2 + gated residual + clip(+-10)
    gemm_kernel<3><<<gemmD, blk, 0, stream>>>(mid16, w2T, b2 + i * Dsz, hbuf, nullptr,
                                              Dsz, Dff, gate2, i, 1.0f);
  }

  // final LN (fp32 out), masked mean pool, heads
  ln_kernel<false><<<Mrows / 8, blk, 0, stream>>>(hbuf, fng, fnb, nullptr, tmp);
  pool_kernel<<<(Bsz * Dsz) / 256, blk, 0, stream>>>(tmp, x, meanB);
  heads_kernel<<<1, 128, 0, stream>>>(meanB,
                                      bhg, bhb, bhw, bhbias,
                                      ahg, ahb, ahw, ahbias,
                                      chg, chb, chw, chbias, outp);
}
